// BoundaryPredictor2_51427938402596
// MI455X (gfx1250) — matrix operand order repaired and checked by reading, not compiled, not run on hardware
//
#include <hip/hip_runtime.h>
#include <stdint.h>

// ---------------------------------------------------------------------------
// CDNA5 (gfx1250) WMMA / TDM types
// ---------------------------------------------------------------------------
typedef __attribute__((ext_vector_type(16))) __bf16 v16bf;
typedef __attribute__((ext_vector_type(8)))  __bf16 bf16x8;
typedef __attribute__((ext_vector_type(8)))  float  v8f;
typedef __attribute__((ext_vector_type(4)))  unsigned int u32x4;
typedef __attribute__((ext_vector_type(8)))  int    i32x8;
typedef __attribute__((ext_vector_type(4)))  int    i32x4;

#define B_   4
#define L_   1024
#define D_   512
#define H_   8
#define HD_  64
#define M_   (B_ * L_)        // 4096 token rows
#define LDS_ROW_B 1056       // 1024B weight row + 32B TDM pad (bank-conflict avoidance)

// ---------------------------------------------------------------------------
// float -> bf16 conversion (weights, done every call; deterministic)
// ---------------------------------------------------------------------------
__global__ __launch_bounds__(256) void k_cvt_bf16(const float* __restrict__ src,
                                                  __bf16* __restrict__ dst, int n) {
    int i = blockIdx.x * 256 + threadIdx.x;
    if (i < n) dst[i] = (__bf16)src[i];
}

// ---------------------------------------------------------------------------
// Per-token prep: nrm = l2norm(hidden_row) (f32 + bf16), hn = LayerNorm (bf16)
// ---------------------------------------------------------------------------
__global__ __launch_bounds__(256) void k_prep(const float* __restrict__ hidden,
                                              const float* __restrict__ ln_g,
                                              const float* __restrict__ ln_b,
                                              float* __restrict__ nrmF,
                                              __bf16* __restrict__ nrmH,
                                              __bf16* __restrict__ hnH) {
    const int tok = blockIdx.x;
    const float* x = hidden + (size_t)tok * D_;
    const int t = threadIdx.x;
    float v0 = x[t], v1 = x[t + 256];
    float sum = v0 + v1;
    float ssq = v0 * v0 + v1 * v1;
    #pragma unroll
    for (int m = 16; m >= 1; m >>= 1) {
        sum += __shfl_xor(sum, m, 32);
        ssq += __shfl_xor(ssq, m, 32);
    }
    __shared__ float ws1[8], ws2[8];
    int wave = t >> 5, lane = t & 31;
    if (lane == 0) { ws1[wave] = sum; ws2[wave] = ssq; }
    __syncthreads();
    if (t == 0) {
        float a = 0.f, b = 0.f;
        #pragma unroll
        for (int i = 0; i < 8; i++) { a += ws1[i]; b += ws2[i]; }
        ws1[0] = a; ws2[0] = b;
    }
    __syncthreads();
    const float tsum = ws1[0], tssq = ws2[0];
    const float inv_n = 1.f / fmaxf(sqrtf(tssq), 1e-8f);
    const float mu    = tsum * (1.f / (float)D_);
    const float var   = tssq * (1.f / (float)D_) - mu * mu;
    const float rstd  = rsqrtf(var + 1e-5f);

    size_t base = (size_t)tok * D_;
    float n0 = v0 * inv_n, n1 = v1 * inv_n;
    nrmF[base + t]       = n0;
    nrmF[base + t + 256] = n1;
    nrmH[base + t]       = (__bf16)n0;
    nrmH[base + t + 256] = (__bf16)n1;
    hnH[base + t]        = (__bf16)((v0 - mu) * rstd * ln_g[t] + ln_b[t]);
    hnH[base + t + 256]  = (__bf16)((v1 - mu) * rstd * ln_g[t + 256] + ln_b[t + 256]);
}

// ---------------------------------------------------------------------------
// Row l2norm of a f32 matrix -> bf16 (for l2norm(mlp(x)+x))
// ---------------------------------------------------------------------------
__global__ __launch_bounds__(256) void k_rownorm(const float* __restrict__ in,
                                                 __bf16* __restrict__ out) {
    const int tok = blockIdx.x;
    const float* x = in + (size_t)tok * D_;
    const int t = threadIdx.x;
    float v0 = x[t], v1 = x[t + 256];
    float ssq = v0 * v0 + v1 * v1;
    #pragma unroll
    for (int m = 16; m >= 1; m >>= 1) ssq += __shfl_xor(ssq, m, 32);
    __shared__ float ws2[8];
    int wave = t >> 5, lane = t & 31;
    if (lane == 0) ws2[wave] = ssq;
    __syncthreads();
    if (t == 0) {
        float b = 0.f;
        #pragma unroll
        for (int i = 0; i < 8; i++) b += ws2[i];
        ws2[0] = b;
    }
    __syncthreads();
    const float inv_n = 1.f / fmaxf(sqrtf(ws2[0]), 1e-8f);
    size_t base = (size_t)tok * D_;
    out[base + t]       = (__bf16)(v0 * inv_n);
    out[base + t + 256] = (__bf16)(v1 * inv_n);
}

// ---------------------------------------------------------------------------
// Tensor Data Mover: DMA a 64-row x 512-col bf16 tile (rows contiguous,
// row stride 512 elems) from global into LDS, inserting 32B padding after
// every 1024B row (pad_interval code 7 = 256 DWORDs, pad_amount code 7 = 8
// DWORDs) so LDS rows are strided LDS_ROW_B bytes.
// D# layout per CDNA5 ISA ch.8 (group0 128b, group1 256b; groups 2/3 zero).
// ---------------------------------------------------------------------------
__device__ __forceinline__ void tdm_load_b_tile(const __bf16* gsrc, uint32_t lds_off) {
    uint64_t ga = (uint64_t)(uintptr_t)gsrc;
    u32x4 g0;
    g0[0] = 1u;                                   // count=1 (valid user D#)
    g0[1] = lds_off;                              // lds_addr
    g0[2] = (uint32_t)ga;                         // global_addr[31:0]
    g0[3] = (uint32_t)((ga >> 32) & 0x01FFFFFFull) | (2u << 30); // addr[56:32], type=2
    i32x8 g1;
    g1[0] = (int)((1u << 16)                      // data_size = 1 -> 2 bytes
                | (1u << 20)                      // pad_enable
                | (7u << 22)                      // pad_interval: 256 DWORDs (one row)
                | (7u << 25));                    // pad_amount:   8 DWORDs (32B)
    g1[1] = (int)(512u << 16);                    // tensor_dim0[15:0] = 512
    g1[2] = (int)(64u  << 16);                    // tensor_dim1[15:0] = 64
    g1[3] = (int)(512u << 16);                    // tile_dim0 = 512
    g1[4] = 64;                                   // tile_dim1 = 64, tile_dim2 = 0
    g1[5] = 512;                                  // tensor_dim0_stride = 512 elems
    g1[6] = 0;
    g1[7] = 0;
    i32x4 z4 = {0, 0, 0, 0};
#if __clang_major__ >= 23
    i32x8 z8 = {0, 0, 0, 0, 0, 0, 0, 0};
    __builtin_amdgcn_tensor_load_to_lds(g0, g1, z4, z4, z8, 0);
#else
    __builtin_amdgcn_tensor_load_to_lds(g0, g1, z4, z4, 0);
#endif
}

// ---------------------------------------------------------------------------
// WMMA bf16 GEMM:  Y[M x 512] = A[M x 512] @ W^T  (W is [out=512,in=512] row-major)
// Block = 256 threads = 8 waves; block tile = 128(M) x 64(N); wave tile = 16 x 64.
// B tile (64 x 512 bf16) staged once per block into LDS by the TDM; A streams
// from global. Epilogue: optional bias / f32 residual / exact GELU; f32 or bf16 out.
// ---------------------------------------------------------------------------
__global__ __launch_bounds__(256) void k_gemm_bf16(const __bf16* __restrict__ A,
                                                   const __bf16* __restrict__ W,
                                                   const float* __restrict__ bias,
                                                   const float* __restrict__ resid,
                                                   float* __restrict__ outF,
                                                   __bf16* __restrict__ outH,
                                                   int do_gelu) {
    const int K = D_;
    const int wave = threadIdx.x >> 5;
    const int lane = threadIdx.x & 31;
    const int half = lane >> 4;     // K-half selector (this kernel's contiguous k = 16*half + s, same for A and B)
    const int mn   = lane & 15;     // A: row-in-tile, B/D: col-in-tile
    const int rowBase = blockIdx.x * 128 + wave * 16;
    const int colBase = blockIdx.y * 64;

    __shared__ __align__(64) uint8_t Bs[64 * LDS_ROW_B];   // 66 KB of 320 KB WGP LDS

    // One TDM issue per block, behind a *scalar* branch (TDM ignores EXEC, so
    // an exec-masked region would not suppress issue; readfirstlane-compare
    // forces an s_cmp/s_cbranch scalar skip for waves 1..7).
    if (__builtin_amdgcn_readfirstlane((int)threadIdx.x) < 32) {
        tdm_load_b_tile(W + (size_t)colBase * K, (uint32_t)(uintptr_t)&Bs[0]);
        __builtin_amdgcn_s_wait_tensorcnt(0);
    }
    __syncthreads();

    v8f acc[4] = {v8f{}, v8f{}, v8f{}, v8f{}};
    const __bf16* arow = A + (size_t)(rowBase + mn) * K;

    for (int k0 = 0; k0 < K; k0 += 32) {
        // A fragment: row mn, K chunk [k0+half*16, +16)
        union { v16bf v; bf16x8 h[2]; } ua;
        ua.h[0] = *(const bf16x8*)(arow + k0 + half * 16);
        ua.h[1] = *(const bf16x8*)(arow + k0 + half * 16 + 8);
        #pragma unroll
        for (int tI = 0; tI < 4; tI++) {
            // B fragment from LDS: column (tI*16+mn) of the tile, K chunk
            // [k0 + half*16, +16) -> 32B ds_load (32B-aligned, padded rows)
            const v16bf bfrag = *(const v16bf*)(
                Bs + (size_t)(tI * 16 + mn) * LDS_ROW_B + (size_t)(k0 + half * 16) * 2);
            acc[tI] = __builtin_amdgcn_wmma_f32_16x16x32_bf16(
                false, ua.v, false, bfrag, (short)0, acc[tI], false, false);
        }
    }

    #pragma unroll
    for (int tI = 0; tI < 4; tI++) {
        const int col = colBase + tI * 16 + mn;
        const float bv = bias ? bias[col] : 0.f;
        #pragma unroll
        for (int r = 0; r < 8; r++) {
            const int row = rowBase + half * 8 + r;
            float v = acc[tI][r] + bv;
            const size_t idx = (size_t)row * K + col;
            if (resid) v += resid[idx];
            if (do_gelu) v = 0.5f * v * (1.f + erff(v * 0.70710678118654752f)); // exact GELU
            if (outF) outF[idx] = v;
            if (outH) outH[idx] = (__bf16)v;
        }
    }
}

// ---------------------------------------------------------------------------
// Adjacent-token cosine: cos[b,l] = dot(Q[b,l], Kh[b,l+1]), l in [0, L-1)
// ---------------------------------------------------------------------------
__global__ __launch_bounds__(256) void k_cos(const float* __restrict__ Q,
                                             const float* __restrict__ Kh,
                                             float* __restrict__ cosA) {
    const int nt = B_ * (L_ - 1);
    const int t = blockIdx.x * 8 + (threadIdx.x >> 5);
    if (t >= nt) return;
    const int lane = threadIdx.x & 31;
    const int b = t / (L_ - 1), l = t % (L_ - 1);
    const float* q = Q  + ((size_t)b * L_ + l)     * D_;
    const float* k = Kh + ((size_t)b * L_ + l + 1) * D_;
    float s = 0.f;
    for (int d = lane; d < D_; d += 32) s += q[d] * k[d];
    #pragma unroll
    for (int m = 16; m >= 1; m >>= 1) s += __shfl_xor(s, m, 32);
    if (lane == 0) cosA[t] = s;
}

// ---------------------------------------------------------------------------
// Boundary logic per batch: probs -> hb -> exclusive cumsum seg_id ->
// contiguous segment ranges under the length mask. One thread per batch.
// ---------------------------------------------------------------------------
__global__ void k_boundary(const float* __restrict__ cosA,
                           const float* __restrict__ lengths,
                           const float* __restrict__ sim_bias,
                           int* __restrict__ segid,
                           int* __restrict__ segst,
                           int* __restrict__ segen) {
    const int b = threadIdx.x;
    if (b >= B_) return;
    const float sb = sim_bias[0];
    const float len = lengths[b];
    const int actual = (int)(len * (float)(L_ + 1));
    int valid = actual - 1;
    valid = valid < 0 ? 0 : (valid > L_ ? L_ : valid);
    const int plen = (int)(len * (float)L_);

    int sumhb = 0;
    for (int l = 0; l < L_; l++) {
        float p = 0.f;
        if (l < L_ - 1) {
            float c = cosA[b * (L_ - 1) + l];
            p = 0.5f * (1.f - (c + sb));
            p = fminf(fmaxf(p, 0.f), 1.f);
        }
        int h = (p > 0.5f) ? 1 : 0;          // hard
        if (l >= valid) h = 0;               // vmask
        if (valid < L_ && l == valid) h = 1; // forced boundary at first pad
        segid[b * L_ + l] = h;               // stash hb temporarily
        sumhb += h;
    }
    if (sumhb == 0) {                        // emergency boundary
        int eidx = valid < (L_ - 1) ? valid : (L_ - 1);
        segid[b * L_ + eidx] = 1;
    }
    int c = 0;                               // seg_id = exclusive cumsum(hb)
    for (int l = 0; l < L_; l++) {
        int h = segid[b * L_ + l];
        segid[b * L_ + l] = c;
        c += h;
    }
    for (int s = 0; s < L_; s++) { segst[b * L_ + s] = -1; segen[b * L_ + s] = -1; }
    for (int l = 0; l < plen; l++) {         // len_mask truncation
        int s = segid[b * L_ + l];
        if (segst[b * L_ + s] < 0) segst[b * L_ + s] = l;
        segen[b * L_ + s] = l + 1;
    }
}

// ---------------------------------------------------------------------------
// Segment attention pooling. One block per (b, s); one wave per head.
// Two-pass masked softmax with scores staged in LDS; empty segment -> zeros.
// ---------------------------------------------------------------------------
__global__ __launch_bounds__(256) void k_pool(const float* __restrict__ Kp,
                                              const float* __restrict__ Vp,
                                              const float* __restrict__ lq,
                                              const int* __restrict__ segst,
                                              const int* __restrict__ segen,
                                              __bf16* __restrict__ pooledH) {
    const int b = blockIdx.x / L_;
    const int s = blockIdx.x % L_;
    const int h = threadIdx.x >> 5;
    const int lane = threadIdx.x & 31;
    __shared__ float sc[H_ * L_];   // 32 KB

    const size_t outoff = ((size_t)b * L_ + s) * D_ + h * HD_;
    const int st = segst[b * L_ + s];
    const int en = segen[b * L_ + s];
    if (st < 0) {
        pooledH[outoff + lane]      = (__bf16)0.f;
        pooledH[outoff + lane + 32] = (__bf16)0.f;
        return;
    }
    const float q0 = lq[h * HD_ + lane];
    const float q1 = lq[h * HD_ + lane + 32];
    const float scale = 0.125f;  // 64^-0.5
    const float* kb = Kp + (size_t)b * L_ * D_ + h * HD_;
    const float* vb = Vp + (size_t)b * L_ * D_ + h * HD_;

    float mx = -3.0e38f;
    for (int l = st; l < en; l++) {
        float p = q0 * kb[(size_t)l * D_ + lane] + q1 * kb[(size_t)l * D_ + lane + 32];
        #pragma unroll
        for (int m = 16; m >= 1; m >>= 1) p += __shfl_xor(p, m, 32);
        p *= scale;                          // all lanes hold the full dot
        if (lane == 0) sc[h * L_ + (l - st)] = p;
        mx = fmaxf(mx, p);
    }
    float denom = 0.f;
    for (int i = lane; i < en - st; i += 32) denom += expf(sc[h * L_ + i] - mx);
    #pragma unroll
    for (int m = 16; m >= 1; m >>= 1) denom += __shfl_xor(denom, m, 32);
    const float inv = 1.f / denom;

    float a0 = 0.f, a1 = 0.f;
    for (int l = st; l < en; l++) {
        float w = expf(sc[h * L_ + (l - st)] - mx) * inv;
        a0 += w * vb[(size_t)l * D_ + lane];
        a1 += w * vb[(size_t)l * D_ + lane + 32];
    }
    pooledH[outoff + lane]      = (__bf16)a0;
    pooledH[outoff + lane + 32] = (__bf16)a1;
}

// ---------------------------------------------------------------------------
// Launch
// ---------------------------------------------------------------------------
extern "C" void kernel_launch(void* const* d_in, const int* in_sizes, int n_in,
                              void* d_out, int out_size, void* d_ws, size_t ws_size,
                              hipStream_t stream) {
    const float* hidden  = (const float*)d_in[0];
    const float* lengths = (const float*)d_in[1];
    const float* W1      = (const float*)d_in[2];
    const float* b1      = (const float*)d_in[3];
    const float* W2      = (const float*)d_in[4];
    const float* b2      = (const float*)d_in[5];
    const float* Wq      = (const float*)d_in[6];
    const float* Wk      = (const float*)d_in[7];
    const float* simb    = (const float*)d_in[8];
    const float* lq      = (const float*)d_in[9];
    const float* Wpk     = (const float*)d_in[10];
    const float* Wpv     = (const float*)d_in[11];
    const float* Wpo     = (const float*)d_in[12];
    const float* ln_g    = (const float*)d_in[13];
    const float* ln_b    = (const float*)d_in[14];

    const size_t MB = 1ull << 20;
    uint8_t* w = (uint8_t*)d_ws;
    // f32 buffers (8 MB each, with deliberate reuse across phases)
    float*  nrmF = (float*)(w + 0 * MB);     // until GEMM2 epilogue
    float*  T2f  = (float*)(w + 8 * MB);     // mlp2 out + residual
    float*  Qf   = (float*)(w + 16 * MB);
    float*  Kf   = (float*)(w + 24 * MB);
    float*  KpF  = (float*)(w + 0 * MB);     // reuse nrmF
    float*  VpF  = (float*)(w + 8 * MB);     // reuse T2f
    // bf16 buffers (4 MB each)
    __bf16* nrmH  = (__bf16*)(w + 32 * MB);
    __bf16* hnH   = (__bf16*)(w + 36 * MB);
    __bf16* T1H   = (__bf16*)(w + 40 * MB);
    __bf16* RnH   = (__bf16*)(w + 44 * MB);
    __bf16* poolH = (__bf16*)(w + 32 * MB);  // reuse nrmH
    // bf16 weights: W1,W2,Wq,Wk,Wpk,Wpv,Wpo (512 KB each)
    __bf16* Wbf   = (__bf16*)(w + 48 * MB);
    const int WN = D_ * D_;
    __bf16 *W1h = Wbf, *W2h = Wbf + WN, *Wqh = Wbf + 2 * WN, *Wkh = Wbf + 3 * WN,
           *Wpkh = Wbf + 4 * WN, *Wpvh = Wbf + 5 * WN, *Wpoh = Wbf + 6 * WN;
    // small buffers
    float* cosA = (float*)(w + 52 * MB);
    int* segid  = (int*)(w + 52 * MB + 64 * 1024);
    int* segst  = (int*)(w + 52 * MB + 128 * 1024);
    int* segen  = (int*)(w + 52 * MB + 192 * 1024);

    const dim3 gcv((WN + 255) / 256), bcv(256);
    k_cvt_bf16<<<gcv, bcv, 0, stream>>>(W1,  W1h,  WN);
    k_cvt_bf16<<<gcv, bcv, 0, stream>>>(W2,  W2h,  WN);
    k_cvt_bf16<<<gcv, bcv, 0, stream>>>(Wq,  Wqh,  WN);
    k_cvt_bf16<<<gcv, bcv, 0, stream>>>(Wk,  Wkh,  WN);
    k_cvt_bf16<<<gcv, bcv, 0, stream>>>(Wpk, Wpkh, WN);
    k_cvt_bf16<<<gcv, bcv, 0, stream>>>(Wpv, Wpvh, WN);
    k_cvt_bf16<<<gcv, bcv, 0, stream>>>(Wpo, Wpoh, WN);

    // per-token l2norm + layernorm
    k_prep<<<M_, 256, 0, stream>>>(hidden, ln_g, ln_b, nrmF, nrmH, hnH);

    const dim3 gg(M_ / 128, D_ / 64), bg(256);
    // T1 = gelu(nrm @ W1^T + b1)
    k_gemm_bf16<<<gg, bg, 0, stream>>>(nrmH, W1h, b1, nullptr, nullptr, T1H, 1);
    // T2 = T1 @ W2^T + b2 + nrm   (residual fused)
    k_gemm_bf16<<<gg, bg, 0, stream>>>(T1H, W2h, b2, nrmF, T2f, nullptr, 0);
    // Rn = l2norm(T2)
    k_rownorm<<<M_, 256, 0, stream>>>(T2f, RnH);
    // Q = Rn @ Wq^T ; Kh = Rn @ Wk^T
    k_gemm_bf16<<<gg, bg, 0, stream>>>(RnH, Wqh, nullptr, nullptr, Qf, nullptr, 0);
    k_gemm_bf16<<<gg, bg, 0, stream>>>(RnH, Wkh, nullptr, nullptr, Kf, nullptr, 0);
    // adjacent cosine
    k_cos<<<(B_ * (L_ - 1) + 7) / 8, 256, 0, stream>>>(Qf, Kf, cosA);
    // boundary / segment ranges
    k_boundary<<<1, B_, 0, stream>>>(cosA, lengths, simb, segid, segst, segen);
    // K = hn @ Wpk^T ; V = hn @ Wpv^T
    k_gemm_bf16<<<gg, bg, 0, stream>>>(hnH, Wpkh, nullptr, nullptr, KpF, nullptr, 0);
    k_gemm_bf16<<<gg, bg, 0, stream>>>(hnH, Wpvh, nullptr, nullptr, VpF, nullptr, 0);
    // segment attention pooling
    k_pool<<<B_ * L_, 256, 0, stream>>>(KpF, VpF, lq, segst, segen, poolH);
    // out = pooled @ Wpo^T  (zero rows stay zero: no bias)
    k_gemm_bf16<<<gg, bg, 0, stream>>>(poolH, Wpoh, nullptr, nullptr, (float*)d_out, nullptr, 0);
}
